// SSMixtralBLockSparseTop2MLP_49443663512201
// MI455X (gfx1250) — matrix operand order repaired and checked by reading, not compiled, not run on hardware
//
#include <hip/hip_runtime.h>
#include <stdint.h>

// MI455X / gfx1250 Mixtral expert MLP:
//   x = gather(hidden, idx)  -> bf16
//   h = silu(x@w1^T) * (x@w3^T)   (WMMA bf16, fused)
//   out = (h@w2^T) * routing       (WMMA bf16, f32 out)
// A-tiles (bf16, no conversion) staged with GLOBAL_LOAD_ASYNC_TO_LDS_B128
// (ASYNCcnt); weight tiles converted f32->bf16 in-flight via v_perm_b32.
// Workspace layout: [x_bf16: 4096*4096*2 B][h_bf16: 4096*14336*2 B] ~= 144 MiB

typedef __attribute__((ext_vector_type(16))) __bf16 v16bf;
typedef __attribute__((ext_vector_type(2)))  __bf16 v2bf;
typedef __attribute__((ext_vector_type(8)))  float  v8f;

#define TOKS 4096
#define HID  4096
#define FFN  14336
#define BK   32
#define LK   40   // padded LDS row stride in halves (20 banks -> conflict-free b128)

union FragU { v16bf v; uint4 q[2]; };

// ---- f32 -> bf16x2 pack: 1 VALU op per 2 elements --------------------------
__device__ __forceinline__ unsigned int pk2(float lo, float hi) {
#if __has_builtin(__builtin_amdgcn_cvt_pk_bf16_f32)
    v2bf t = __builtin_amdgcn_cvt_pk_bf16_f32(lo, hi);   // v_cvt_pk_bf16_f32 (RNE)
    return __builtin_bit_cast(unsigned int, t);
#elif __has_builtin(__builtin_amdgcn_perm)
    // v_perm_b32: result = {hi16(hi), hi16(lo)} (truncating bf16 convert)
    return __builtin_amdgcn_perm(__float_as_uint(hi), __float_as_uint(lo),
                                 0x07060302u);
#else
    unsigned int a = __float_as_uint(lo), b = __float_as_uint(hi);
    a += 0x7FFFu + ((a >> 16) & 1u);
    b += 0x7FFFu + ((b >> 16) & 1u);
    return (a >> 16) | (b & 0xFFFF0000u);
#endif
}
__device__ __forceinline__ unsigned short f2bf_s(float f) {
    return (unsigned short)pk2(f, 0.0f);
}
__device__ __forceinline__ uint4 pack8(float4 a, float4 b) {
    uint4 r;
    r.x = pk2(a.x, a.y); r.y = pk2(a.z, a.w);
    r.z = pk2(b.x, b.y); r.w = pk2(b.z, b.w);
    return r;
}

// ---- async global->LDS copy of 32 bytes (2 x b128), per lane ---------------
// the instruction offset is added to both the LDS and the global address
__device__ __forceinline__ void async_copy32(const unsigned short* gsrc,
                                             unsigned short* lds_dst) {
    unsigned int       lv = (unsigned int)(uintptr_t)lds_dst;       // LDS byte addr
    unsigned long long ga = (unsigned long long)(uintptr_t)gsrc;
    asm volatile("global_load_async_to_lds_b128 %0, %1, off\n\t"
                 "global_load_async_to_lds_b128 %0, %1, off offset:16"
                 :: "v"(lv), "v"(ga) : "memory");
}
__device__ __forceinline__ void wait_async0() {
    asm volatile("s_wait_asynccnt 0" ::: "memory");
}

// ---------------------------------------------------------------- gather ----
__global__ __launch_bounds__(256) void gather_bf16_kernel(
    const float* __restrict__ hs, const long long* __restrict__ idx,
    unsigned short* __restrict__ xb) {
    const int n = blockIdx.x;
    const long long src = idx[n];
    const float4* sp = (const float4*)(hs + (size_t)src * HID);
    unsigned short* dp = xb + (size_t)n * HID;
    const int t = threadIdx.x;
#pragma unroll
    for (int i = 0; i < 4; ++i) {
        const int e4 = t + i * 256;           // float4 index 0..1023
        float4 v = sp[e4];
        uint2 p;
        p.x = pk2(v.x, v.y);
        p.y = pk2(v.z, v.w);
        *(uint2*)(dp + e4 * 4) = p;
    }
}

// ------------------------------------------------------- GEMM1: gate+up ----
// C tile 128(M) x 64(N), BK=32; 8 waves in 4(M) x 2(N), wave tile 32x32.
__global__ __launch_bounds__(256) void gemm1_silu_kernel(
    const unsigned short* __restrict__ xb,
    const float* __restrict__ w1,
    const float* __restrict__ w3,
    unsigned short* __restrict__ hb) {
    __shared__ unsigned short Xs[2][128 * LK];
    __shared__ unsigned short W1s[2][64 * LK];
    __shared__ unsigned short W3s[2][64 * LK];

    const int tid = threadIdx.x;
    const int bn = blockIdx.x;   // FFN / 64
    const int bm = blockIdx.y;   // TOKS / 128

    // staging: X tile 128x32 halves (16 halves/thread), W tiles 64x32 f32 (8/thread)
    const int xrow = tid >> 1, xch = (tid & 1) << 4;
    const int wrow = tid >> 2, wch = (tid & 3) << 3;
    const unsigned short* xg = xb + (size_t)(bm * 128 + xrow) * HID + xch;
    const float* w1g = w1 + (size_t)(bn * 64 + wrow) * HID + wch;
    const float* w3g = w3 + (size_t)(bn * 64 + wrow) * HID + wch;

    uint4 wp1, wp3;
    auto stage_x_async = [&](int buf, int k0) {
        async_copy32(xg + k0, &Xs[buf][xrow * LK + xch]);
    };
    auto load_regs = [&](int k0) {
        __builtin_prefetch(w1g + k0 + BK, 0, 1);
        __builtin_prefetch(w3g + k0 + BK, 0, 1);
        float4 a = *(const float4*)(w1g + k0);
        float4 b = *(const float4*)(w1g + k0 + 4);
        wp1 = pack8(a, b);
        a = *(const float4*)(w3g + k0);
        b = *(const float4*)(w3g + k0 + 4);
        wp3 = pack8(a, b);
    };
    auto store_lds = [&](int buf) {
        *(uint4*)&W1s[buf][wrow * LK + wch] = wp1;
        *(uint4*)&W3s[buf][wrow * LK + wch] = wp3;
    };

    const int lane = tid & 31;
    const int wv = tid >> 5;
    const int wmo = (wv >> 1) << 5;   // 0,32,64,96
    const int wno = (wv & 1) << 5;    // 0,32
    const int r  = lane & 15;
    const int ak = (lane >> 4) << 4;  // A frag: K contiguous 16
    const int bk = (lane >> 4) << 4;  // B frag: K contiguous 16

    v8f zero;
#pragma unroll
    for (int i = 0; i < 8; ++i) zero[i] = 0.0f;
    v8f acc1[2][2], acc3[2][2];
#pragma unroll
    for (int m = 0; m < 2; ++m)
#pragma unroll
        for (int n = 0; n < 2; ++n) { acc1[m][n] = zero; acc3[m][n] = zero; }

    stage_x_async(0, 0);
    load_regs(0);
    store_lds(0);
    wait_async0();
    __syncthreads();

    const int NK = HID / BK;
    for (int kt = 0; kt < NK; ++kt) {
        const int cur = kt & 1;
        const bool pf = (kt + 1) < NK;
        if (pf) {
            stage_x_async(cur ^ 1, (kt + 1) * BK);
            load_regs((kt + 1) * BK);
        }

        FragU a[2], b1f[2], b3f[2];
#pragma unroll
        for (int m = 0; m < 2; ++m) {
            const uint4* p = (const uint4*)&Xs[cur][(wmo + m * 16 + r) * LK + ak];
            a[m].q[0] = p[0]; a[m].q[1] = p[1];
        }
#pragma unroll
        for (int n = 0; n < 2; ++n) {
            const uint4* p1 = (const uint4*)&W1s[cur][(wno + n * 16 + r) * LK + bk];
            b1f[n].q[0] = p1[0]; b1f[n].q[1] = p1[1];
            const uint4* p3 = (const uint4*)&W3s[cur][(wno + n * 16 + r) * LK + bk];
            b3f[n].q[0] = p3[0]; b3f[n].q[1] = p3[1];
        }
#pragma unroll
        for (int m = 0; m < 2; ++m)
#pragma unroll
            for (int n = 0; n < 2; ++n) {
                acc1[m][n] = __builtin_amdgcn_wmma_f32_16x16x32_bf16(
                    false, a[m].v, false, b1f[n].v, (short)0, acc1[m][n], false, false);
                acc3[m][n] = __builtin_amdgcn_wmma_f32_16x16x32_bf16(
                    false, a[m].v, false, b3f[n].v, (short)0, acc3[m][n], false, false);
            }
        if (pf) store_lds(cur ^ 1);
        wait_async0();
        __syncthreads();
    }

    // epilogue: h = silu(g) * u  -> bf16
    const int rbase = bm * 128 + wmo + ((lane < 16) ? 0 : 8);
    const int cbase = bn * 64 + wno + r;
#pragma unroll
    for (int m = 0; m < 2; ++m)
#pragma unroll
        for (int n = 0; n < 2; ++n) {
            const int col = cbase + n * 16;
#pragma unroll
            for (int j = 0; j < 8; ++j) {
                const float g = acc1[m][n][j];
                const float u = acc3[m][n][j];
                const float s = g / (1.0f + __expf(-g));
                const int row = rbase + m * 16 + j;
                hb[(size_t)row * FFN + col] = f2bf_s(s * u);
            }
        }
}

// --------------------------------------------------------- GEMM2: down ----
__global__ __launch_bounds__(256) void gemm2_down_kernel(
    const unsigned short* __restrict__ hbuf,
    const float* __restrict__ w2,
    const float* __restrict__ rw,
    float* __restrict__ out) {
    __shared__ unsigned short Hs[2][128 * LK];
    __shared__ unsigned short W2s[2][64 * LK];

    const int tid = threadIdx.x;
    const int bn = blockIdx.x;   // HID / 64
    const int bm = blockIdx.y;   // TOKS / 128

    const int xrow = tid >> 1, xch = (tid & 1) << 4;
    const int wrow = tid >> 2, wch = (tid & 3) << 3;
    const unsigned short* xg = hbuf + (size_t)(bm * 128 + xrow) * FFN + xch;
    const float* w2g = w2 + (size_t)(bn * 64 + wrow) * FFN + wch;

    uint4 wp2;
    auto stage_h_async = [&](int buf, int k0) {
        async_copy32(xg + k0, &Hs[buf][xrow * LK + xch]);
    };
    auto load_regs = [&](int k0) {
        __builtin_prefetch(w2g + k0 + BK, 0, 1);
        float4 a = *(const float4*)(w2g + k0);
        float4 b = *(const float4*)(w2g + k0 + 4);
        wp2 = pack8(a, b);
    };
    auto store_lds = [&](int buf) {
        *(uint4*)&W2s[buf][wrow * LK + wch] = wp2;
    };

    const int lane = tid & 31;
    const int wv = tid >> 5;
    const int wmo = (wv >> 1) << 5;
    const int wno = (wv & 1) << 5;
    const int r  = lane & 15;
    const int ak = (lane >> 4) << 4;
    const int bk = (lane >> 4) << 4;

    v8f zero;
#pragma unroll
    for (int i = 0; i < 8; ++i) zero[i] = 0.0f;
    v8f acc[2][2];
#pragma unroll
    for (int m = 0; m < 2; ++m)
#pragma unroll
        for (int n = 0; n < 2; ++n) acc[m][n] = zero;

    stage_h_async(0, 0);
    load_regs(0);
    store_lds(0);
    wait_async0();
    __syncthreads();

    const int NK = FFN / BK;   // 448
    for (int kt = 0; kt < NK; ++kt) {
        const int cur = kt & 1;
        const bool pf = (kt + 1) < NK;
        if (pf) {
            stage_h_async(cur ^ 1, (kt + 1) * BK);
            load_regs((kt + 1) * BK);
        }

        FragU a[2], bf[2];
#pragma unroll
        for (int m = 0; m < 2; ++m) {
            const uint4* p = (const uint4*)&Hs[cur][(wmo + m * 16 + r) * LK + ak];
            a[m].q[0] = p[0]; a[m].q[1] = p[1];
        }
#pragma unroll
        for (int n = 0; n < 2; ++n) {
            const uint4* p = (const uint4*)&W2s[cur][(wno + n * 16 + r) * LK + bk];
            bf[n].q[0] = p[0]; bf[n].q[1] = p[1];
        }
#pragma unroll
        for (int m = 0; m < 2; ++m)
#pragma unroll
            for (int n = 0; n < 2; ++n)
                acc[m][n] = __builtin_amdgcn_wmma_f32_16x16x32_bf16(
                    false, a[m].v, false, bf[n].v, (short)0, acc[m][n], false, false);
        if (pf) store_lds(cur ^ 1);
        wait_async0();
        __syncthreads();
    }

    // epilogue: scale by routing weight, f32 out
    const int rbase = bm * 128 + wmo + ((lane < 16) ? 0 : 8);
    const int cbase = bn * 64 + wno + r;
#pragma unroll
    for (int m = 0; m < 2; ++m)
#pragma unroll
        for (int n = 0; n < 2; ++n) {
            const int col = cbase + n * 16;
#pragma unroll
            for (int j = 0; j < 8; ++j) {
                const int row = rbase + m * 16 + j;
                out[(size_t)row * HID + col] = acc[m][n][j] * rw[row];
            }
        }
}

// ----------------------------------------------------------------- launch --
extern "C" void kernel_launch(void* const* d_in, const int* in_sizes, int n_in,
                              void* d_out, int out_size, void* d_ws, size_t ws_size,
                              hipStream_t stream) {
    (void)in_sizes; (void)n_in; (void)out_size; (void)ws_size;
    const float*     hs  = (const float*)d_in[0];
    const long long* idx = (const long long*)d_in[1];   // int64 per reference
    const float*     rw  = (const float*)d_in[2];
    const float*     w1  = (const float*)d_in[3];
    const float*     w2  = (const float*)d_in[4];
    const float*     w3  = (const float*)d_in[5];
    float* out = (float*)d_out;

    unsigned short* xb = (unsigned short*)d_ws;
    unsigned short* hb = (unsigned short*)((char*)d_ws + (size_t)TOKS * HID * 2);

    gather_bf16_kernel<<<TOKS, 256, 0, stream>>>(hs, idx, xb);

    dim3 g1(FFN / 64, TOKS / 128);
    gemm1_silu_kernel<<<g1, 256, 0, stream>>>(xb, w1, w3, hb);

    dim3 g2(HID / 64, TOKS / 128);
    gemm2_down_kernel<<<g2, 256, 0, stream>>>(hb, w2, rw, out);
}
